// DualAttentionRNN_21964462752115
// MI455X (gfx1250) — matrix operand order repaired and checked by reading, not compiled, not run on hardware
//
#include <hip/hip_runtime.h>

// DA-RNN (dual-attention RNN) for MI455X / gfx1250 (wave32, WMMA bf16).
// B=4096, N=256, T=64, H=64, M=64

#define B_ 4096
#define N_ 256
#define T_ 64
#define H_ 64
#define M_ 64

typedef __attribute__((ext_vector_type(16))) __bf16 v16bf;
typedef __attribute__((ext_vector_type(8)))  float  v8f;

// ---- WMMA fragment index maps (CDNA5 ISA, 16-bit layouts) ----
// A (16xK, M along lanes 0..15 repeated): element e of lane -> K offset
__device__ __forceinline__ int kmapA(int e, int half) { return e + (half << 3) + (e & 8); }
// B (Kx16, N along lanes 0..15): same K order as A (lanes 0-15 K=0..7,16..23; lanes 16-31 K=8..15,24..31)
__device__ __forceinline__ int kmapB(int e, int half) { return e + (half << 3) + (e & 8); }

__device__ __forceinline__ v8f wmma_bf16(v16bf a, v16bf b, v8f c) {
  return __builtin_amdgcn_wmma_f32_16x16x32_bf16(false, a, false, b, (short)0, c,
                                                 false, false);
}

__device__ __forceinline__ v8f v8f_zero() {
  v8f z;
#pragma unroll
  for (int i = 0; i < 8; ++i) z[i] = 0.f;
  return z;
}

__device__ __forceinline__ float sigmoidf_(float x) { return 1.f / (1.f + __expf(-x)); }

// B-frag from global f32 weight used TRANSPOSED: Bmat[k][n] = W[n*ldw + k]
__device__ __forceinline__ v16bf load_bT_global(const float* W, int ldw, int n0,
                                                int k0, int lane) {
  const int n = n0 + (lane & 15), half = lane >> 4;
  v16bf b;
#pragma unroll
  for (int e = 0; e < 16; ++e) b[e] = (__bf16)W[n * ldw + k0 + kmapB(e, half)];
  return b;
}
// B-frag from global f32 weight, row-major: Bmat[k][n] = W[k*ldw + n]
__device__ __forceinline__ v16bf load_b_global(const float* W, int ldw, int n0,
                                               int k0, int lane) {
  const int n = n0 + (lane & 15), half = lane >> 4;
  v16bf b;
#pragma unroll
  for (int e = 0; e < 16; ++e) b[e] = (__bf16)W[(k0 + kmapB(e, half)) * ldw + n];
  return b;
}
// A-frag from LDS bf16 tile, row = lane&15
__device__ __forceinline__ v16bf load_a_lds(const __bf16* S, int lda, int k0, int lane) {
  const int m = lane & 15, half = lane >> 4;
  v16bf a;
#pragma unroll
  for (int e = 0; e < 16; ++e) a[e] = S[m * lda + k0 + kmapA(e, half)];
  return a;
}
// B-frag from LDS bf16 tile, row-major Bmat[k][n]
__device__ __forceinline__ v16bf load_b_lds(const __bf16* S, int ldb, int n0, int k0,
                                            int lane) {
  const int n = n0 + (lane & 15), half = lane >> 4;
  v16bf b;
#pragma unroll
  for (int e = 0; e < 16; ++e) b[e] = S[(k0 + kmapB(e, half)) * ldb + n];
  return b;
}

// ---------------------------------------------------------------------------
// Kernel 1: transpose driving_series [b][n][t] f32 -> drvT [b][t][n] bf16
// ---------------------------------------------------------------------------
__global__ __launch_bounds__(256) void k_transpose_drv(const float* __restrict__ drv,
                                                       __bf16* __restrict__ drvT) {
  __shared__ float tile[64][65];
  const int b = blockIdx.x >> 2, nblk = blockIdx.x & 3;
  const int tid = threadIdx.x;
  const float* src = drv + ((long)b * N_ + nblk * 64) * T_;
  for (int idx = tid; idx < 4096; idx += 256) {
    const int n = idx >> 6, t = idx & 63;
    tile[n][t] = src[idx];
  }
  __syncthreads();
  __bf16* dst = drvT + (long)b * T_ * N_ + nblk * 64;
  for (int idx = tid; idx < 4096; idx += 256) {
    const int t = idx >> 6, n = idx & 63;
    dst[(long)t * N_ + n] = (__bf16)tile[n][t];
  }
}

// ---------------------------------------------------------------------------
// Kernel 2/4: C[rows x 64] = A[rows x 64] @ W[64 x 64]  (bf16 WMMA, f32 acc)
// One WG (128 thr = 4 waves) computes a 64-row x 64-col tile.
// ---------------------------------------------------------------------------
template <bool OUT_BF16>
__global__ __launch_bounds__(128) void k_gemm64(const float* __restrict__ A,
                                                const float* __restrict__ W,
                                                void* __restrict__ C) {
  __shared__ __bf16 sA[64 * 64];
  __shared__ __bf16 sW[64 * 64];
  const int tid = threadIdx.x;
  const long rowBase = (long)blockIdx.x * 64;
  const float* Ab = A + rowBase * 64;
  for (int idx = tid; idx < 4096; idx += 128) {
    sW[idx] = (__bf16)W[idx];
    sA[idx] = (__bf16)Ab[idx];
  }
  __syncthreads();
  const int wave = tid >> 5, lane = tid & 31;
  const int half = lane >> 4, l16 = lane & 15;
  const int m0 = wave * 16;
  v8f acc[4];
#pragma unroll
  for (int nt = 0; nt < 4; ++nt) acc[nt] = v8f_zero();
#pragma unroll
  for (int kt = 0; kt < 2; ++kt) {
    const int k0 = kt * 32;
    const v16bf a = load_a_lds(sA + m0 * 64, 64, k0, lane);
#pragma unroll
    for (int nt = 0; nt < 4; ++nt) {
      const v16bf b = load_b_lds(sW, 64, nt * 16, k0, lane);
      acc[nt] = wmma_bf16(a, b, acc[nt]);
    }
  }
#pragma unroll
  for (int nt = 0; nt < 4; ++nt)
#pragma unroll
    for (int v = 0; v < 8; ++v) {
      const long r = rowBase + m0 + v + 8 * half;
      const int n = nt * 16 + l16;
      if (OUT_BF16)
        ((__bf16*)C)[r * 64 + n] = (__bf16)acc[nt][v];
      else
        ((float*)C)[r * 64 + n] = acc[nt][v];
    }
}

// ---------------------------------------------------------------------------
// Kernel 3: fused encoder scan. 256 thr (8 waves) per WG, 16 batches per WG.
// Weights live in register-resident WMMA B-fragments across all 64 steps.
// ---------------------------------------------------------------------------
__global__ __launch_bounds__(256) void k_encoder(
    const __bf16* __restrict__ pre_in, const __bf16* __restrict__ drvT,
    const float* __restrict__ W_in, const float* __restrict__ v_in,
    const float* __restrict__ Wih_e, const float* __restrict__ Whh_e,
    const float* __restrict__ bih_e, const float* __restrict__ bhh_e,
    float* __restrict__ H_out) {
  __shared__ float s_hid[16 * 64];      // hid[b][t]
  __shared__ float s_cx[16 * 64];       // cell state f32
  __shared__ __bf16 s_hc[16 * 128];     // concat(hx, cx) bf16 (GEMM A operand)
  __shared__ __bf16 s_inp[16 * 256];    // attended input (GEMM A operand)
  __shared__ float s_gates[16 * 256];   // gate pre-activations / logits scratch
  __shared__ float s_red[256];          // softmax reductions
  __shared__ float s_vin[64];
  __shared__ float s_bias[256];

  const int tid = threadIdx.x;
  const int wave = tid >> 5, lane = tid & 31;
  const int half = lane >> 4, l16 = lane & 15;
  const long bg0 = (long)blockIdx.x * 16;

  for (int idx = tid; idx < 16 * 64; idx += 256) s_cx[idx] = 0.f;
  for (int idx = tid; idx < 16 * 128; idx += 256) s_hc[idx] = (__bf16)0.f;
  if (tid < 64) s_vin[tid] = v_in[tid];
  s_bias[tid] = bih_e[tid] + bhh_e[tid];

  // Step-invariant weight fragments (held in VGPRs for all 64 steps).
  v16bf win_f[4];  // W_in (128x64) row-major; waves 0-3 own one 16-col tile each
  if (wave < 4) {
#pragma unroll
    for (int kt = 0; kt < 4; ++kt)
      win_f[kt] = load_b_global(W_in, 64, wave * 16, kt * 32, lane);
  }
  v16bf wih_f[2][8];  // Wih_e^T (256k x 256n): each wave owns 2 of 16 n-tiles
  v16bf whh_f[2][2];  // Whh_e^T (64k x 256n)
#pragma unroll
  for (int c = 0; c < 2; ++c) {
    const int n0 = (2 * wave + c) * 16;
#pragma unroll
    for (int kt = 0; kt < 8; ++kt) wih_f[c][kt] = load_bT_global(Wih_e, 256, n0, kt * 32, lane);
#pragma unroll
    for (int kt = 0; kt < 2; ++kt) whh_f[c][kt] = load_bT_global(Whh_e, 64, n0, kt * 32, lane);
  }
  __syncthreads();

  const int b = tid >> 4, q = tid & 15;  // thread -> (batch, 16-way slice)
  const long bg = bg0 + b;
  const __bf16* prow_base = pre_in + (bg * N_ + q * 16) * (long)T_;

  for (int i = 0; i < T_; ++i) {
    // --- phase 1: hid = [hx|cx](16x128) @ W_in(128x64) -> s_hid ---
    if (wave < 4) {
      v8f acc = v8f_zero();
#pragma unroll
      for (int kt = 0; kt < 4; ++kt) {
        const v16bf a = load_a_lds(s_hc, 128, kt * 32, lane);
        acc = wmma_bf16(a, win_f[kt], acc);
      }
#pragma unroll
      for (int v = 0; v < 8; ++v)
        s_hid[(v + 8 * half) * 64 + wave * 16 + l16] = acc[v];
    }
    __syncthreads();

    // --- phase 2: input attention: logits, softmax over N=256, inp ---
    {
      float pmax = -3.0e38f;
      for (int r = 0; r < 16; ++r) {  // 16 rows n = q*16 + r (pre_in is L2-resident bf16)
        const __bf16* pr = prow_base + r * T_;
        float s = 0.f;
#pragma unroll 4
        for (int t = 0; t < 64; ++t)
          s += tanhf((float)pr[t] + s_hid[b * 64 + t]) * s_vin[t];
        s_gates[b * 256 + q * 16 + r] = s;  // stash logits (s_gates free here)
        pmax = fmaxf(pmax, s);
      }
      s_red[tid] = pmax;
      __syncthreads();
      float bmax = -3.0e38f;
#pragma unroll
      for (int u = 0; u < 16; ++u) bmax = fmaxf(bmax, s_red[b * 16 + u]);
      float psum = 0.f;
      for (int r = 0; r < 16; ++r) {
        const float ex = __expf(s_gates[b * 256 + q * 16 + r] - bmax);
        s_gates[b * 256 + q * 16 + r] = ex;
        psum += ex;
      }
      __syncthreads();  // everyone done reading s_red(max)
      s_red[tid] = psum;
      __syncthreads();
      float bsum = 0.f;
#pragma unroll
      for (int u = 0; u < 16; ++u) bsum += s_red[b * 16 + u];
      const float inv = 1.f / bsum;
      const __bf16* xrow = drvT + (bg * T_ + i) * (long)N_ + q * 16;
      for (int r = 0; r < 16; ++r) {
        const float w = s_gates[b * 256 + q * 16 + r] * inv;
        s_inp[b * 256 + q * 16 + r] = (__bf16)(w * (float)xrow[r]);
      }
    }
    __syncthreads();

    // --- phase 3: gates = inp@Wih_e^T + hx@Whh_e^T + bias (WMMA) ---
    {
      v8f acc[2];
#pragma unroll
      for (int c = 0; c < 2; ++c) {
        const float bs = s_bias[(2 * wave + c) * 16 + l16];
#pragma unroll
        for (int v = 0; v < 8; ++v) acc[c][v] = bs;
      }
#pragma unroll
      for (int kt = 0; kt < 8; ++kt) {
        const v16bf a = load_a_lds(s_inp, 256, kt * 32, lane);
        acc[0] = wmma_bf16(a, wih_f[0][kt], acc[0]);
        acc[1] = wmma_bf16(a, wih_f[1][kt], acc[1]);
      }
#pragma unroll
      for (int kt = 0; kt < 2; ++kt) {
        const v16bf a = load_a_lds(s_hc, 128, kt * 32, lane);  // hx half
        acc[0] = wmma_bf16(a, whh_f[0][kt], acc[0]);
        acc[1] = wmma_bf16(a, whh_f[1][kt], acc[1]);
      }
#pragma unroll
      for (int c = 0; c < 2; ++c)
#pragma unroll
        for (int v = 0; v < 8; ++v)
          s_gates[(v + 8 * half) * 256 + (2 * wave + c) * 16 + l16] = acc[c][v];
    }
    __syncthreads();

    // --- phase 4: LSTM activations, state update, H_out store ---
    {
      float* Hrow = H_out + (bg * T_ + i) * (long)H_;
#pragma unroll
      for (int r = 0; r < 4; ++r) {
        const int h = q * 4 + r;
        const float ig = sigmoidf_(s_gates[b * 256 + h]);
        const float fg = sigmoidf_(s_gates[b * 256 + 64 + h]);
        const float gg = tanhf(s_gates[b * 256 + 128 + h]);
        const float og = sigmoidf_(s_gates[b * 256 + 192 + h]);
        const float cn = fg * s_cx[b * 64 + h] + ig * gg;
        const float hn = og * tanhf(cn);
        s_cx[b * 64 + h] = cn;
        s_hc[b * 128 + h] = (__bf16)hn;
        s_hc[b * 128 + 64 + h] = (__bf16)cn;
        Hrow[h] = hn;
      }
    }
    __syncthreads();
  }
}

// ---------------------------------------------------------------------------
// Kernel 5: fused decoder scan + output projection.
// ---------------------------------------------------------------------------
__global__ __launch_bounds__(256) void k_decoder(
    const float* __restrict__ history, const float* __restrict__ pre_h,
    const float* __restrict__ Hout, const float* __restrict__ W_t,
    const float* __restrict__ v_t, const float* __restrict__ Wih_d,
    const float* __restrict__ Whh_d, const float* __restrict__ bih_d,
    const float* __restrict__ bhh_d, const float* __restrict__ Wy,
    const float* __restrict__ by, const float* __restrict__ Wo,
    const float* __restrict__ bo, float* __restrict__ out) {
  __shared__ float s_hid[16 * 64];
  __shared__ float s_hx[16 * 64];
  __shared__ float s_cx[16 * 64];
  __shared__ __bf16 s_hc[16 * 128];
  __shared__ float s_w[16 * 64];   // temporal softmax weights
  __shared__ float s_ci[16 * 64];  // context vector
  __shared__ float s_gates[16 * 256];
  __shared__ float s_red[256];
  __shared__ float s_vt[64];
  __shared__ float s_bias[256];
  __shared__ float s_wihd[256];
  __shared__ float s_wy[66];
  __shared__ float s_yhat[16];

  const int tid = threadIdx.x;
  const int wave = tid >> 5, lane = tid & 31;
  const int half = lane >> 4, l16 = lane & 15;
  const long bg0 = (long)blockIdx.x * 16;

  for (int idx = tid; idx < 16 * 64; idx += 256) { s_hx[idx] = 0.f; s_cx[idx] = 0.f; }
  for (int idx = tid; idx < 16 * 128; idx += 256) s_hc[idx] = (__bf16)0.f;
  if (tid < 64) s_vt[tid] = v_t[tid];
  s_bias[tid] = bih_d[tid] + bhh_d[tid];
  s_wihd[tid] = Wih_d[tid];
  if (tid < 65) s_wy[tid] = Wy[tid];
  if (tid == 0) s_wy[65] = by[0];

  v16bf wt_f[4];  // W_t (128x64) row-major
  if (wave < 4) {
#pragma unroll
    for (int kt = 0; kt < 4; ++kt)
      wt_f[kt] = load_b_global(W_t, 64, wave * 16, kt * 32, lane);
  }
  v16bf whhd_f[2][2];  // Whh_d^T (64k x 256n)
#pragma unroll
  for (int c = 0; c < 2; ++c) {
    const int n0 = (2 * wave + c) * 16;
#pragma unroll
    for (int kt = 0; kt < 2; ++kt) whhd_f[c][kt] = load_bT_global(Whh_d, 64, n0, kt * 32, lane);
  }
  __syncthreads();

  const int b = tid >> 4, q = tid & 15;
  const long bg = bg0 + b;

  for (int i = 0; i < T_; ++i) {
    // --- phase 1: hid = [hx|cx](16x128) @ W_t(128x64) ---
    if (wave < 4) {
      v8f acc = v8f_zero();
#pragma unroll
      for (int kt = 0; kt < 4; ++kt) {
        const v16bf a = load_a_lds(s_hc, 128, kt * 32, lane);
        acc = wmma_bf16(a, wt_f[kt], acc);
      }
#pragma unroll
      for (int v = 0; v < 8; ++v)
        s_hid[(v + 8 * half) * 64 + wave * 16 + l16] = acc[v];
    }
    __syncthreads();

    // --- phase 2: temporal attention softmax over T=64 ---
    {
      float lg[4];
      float pmax = -3.0e38f;
#pragma unroll
      for (int r = 0; r < 4; ++r) {
        const int tt = q * 4 + r;
        const float* ph = pre_h + (bg * T_ + tt) * (long)H_;
        float s = 0.f;
#pragma unroll 4
        for (int h = 0; h < 64; ++h)
          s += tanhf(ph[h] + s_hid[b * 64 + h]) * s_vt[h];
        lg[r] = s;
        pmax = fmaxf(pmax, s);
      }
      s_red[tid] = pmax;
      __syncthreads();
      float bmax = -3.0e38f;
#pragma unroll
      for (int u = 0; u < 16; ++u) bmax = fmaxf(bmax, s_red[b * 16 + u]);
      float psum = 0.f;
#pragma unroll
      for (int r = 0; r < 4; ++r) { lg[r] = __expf(lg[r] - bmax); psum += lg[r]; }
      __syncthreads();
      s_red[tid] = psum;
      __syncthreads();
      float bsum = 0.f;
#pragma unroll
      for (int u = 0; u < 16; ++u) bsum += s_red[b * 16 + u];
      const float inv = 1.f / bsum;
#pragma unroll
      for (int r = 0; r < 4; ++r) s_w[b * 64 + q * 4 + r] = lg[r] * inv;
    }
    __syncthreads();

    // --- phase 3: context c_i[b][h] = sum_t w[b][t] * H_out[b][t][h] ---
#pragma unroll
    for (int r = 0; r < 4; ++r) {
      const int h = q + 16 * r;
      float s = 0.f;
      for (int t = 0; t < 64; ++t)
        s += s_w[b * 64 + t] * Hout[(bg * T_ + t) * (long)H_ + h];
      s_ci[b * 64 + h] = s;
    }
    __syncthreads();

    // --- phase 4: y_hat = [c_i, y_t] @ Wy^T + by (threads 0..15) ---
    if (tid < 16) {
      const int bb = tid;
      float yv = 0.f;
      for (int h = 0; h < 64; ++h) yv += s_ci[bb * 64 + h] * s_wy[h];
      const float yt = history[(bg0 + bb) * T_ + i];
      s_yhat[bb] = yv + yt * s_wy[64] + s_wy[65];
    }

    // --- phase 5: gates = hx @ Whh_d^T + bias (WMMA) ---
    {
      v8f acc[2];
#pragma unroll
      for (int c = 0; c < 2; ++c) {
        const float bs = s_bias[(2 * wave + c) * 16 + l16];
#pragma unroll
        for (int v = 0; v < 8; ++v) acc[c][v] = bs;
      }
#pragma unroll
      for (int kt = 0; kt < 2; ++kt) {
        const v16bf a = load_a_lds(s_hc, 128, kt * 32, lane);  // hx half
        acc[0] = wmma_bf16(a, whhd_f[0][kt], acc[0]);
        acc[1] = wmma_bf16(a, whhd_f[1][kt], acc[1]);
      }
#pragma unroll
      for (int c = 0; c < 2; ++c)
#pragma unroll
        for (int v = 0; v < 8; ++v)
          s_gates[(v + 8 * half) * 256 + (2 * wave + c) * 16 + l16] = acc[c][v];
    }
    __syncthreads();

    // --- phase 6: activations (+ rank-1 y_hat * Wih_d term) ---
    {
#pragma unroll
      for (int r = 0; r < 4; ++r) {
        const int h = q * 4 + r;
        const float yh = s_yhat[b];
        const float ig = sigmoidf_(s_gates[b * 256 + h] + yh * s_wihd[h]);
        const float fg = sigmoidf_(s_gates[b * 256 + 64 + h] + yh * s_wihd[64 + h]);
        const float gg = tanhf(s_gates[b * 256 + 128 + h] + yh * s_wihd[128 + h]);
        const float og = sigmoidf_(s_gates[b * 256 + 192 + h] + yh * s_wihd[192 + h]);
        const float cn = fg * s_cx[b * 64 + h] + ig * gg;
        const float hn = og * tanhf(cn);
        s_cx[b * 64 + h] = cn;
        s_hx[b * 64 + h] = hn;
        s_hc[b * 128 + h] = (__bf16)hn;
        s_hc[b * 128 + 64 + h] = (__bf16)cn;
      }
    }
    __syncthreads();
  }

  // --- final: out[b] = hx @ Wo^T + bo ---
  if (tid < 32) {
    const int bb = tid >> 1, o = tid & 1;
    float s = bo[o];
    for (int h = 0; h < 64; ++h) s += s_hx[bb * 64 + h] * Wo[o * 64 + h];
    out[(bg0 + bb) * 2 + o] = s;
  }
}

// ---------------------------------------------------------------------------
extern "C" void kernel_launch(void* const* d_in, const int* in_sizes, int n_in,
                              void* d_out, int out_size, void* d_ws, size_t ws_size,
                              hipStream_t stream) {
  (void)in_sizes; (void)n_in; (void)out_size;

  const float* history = (const float*)d_in[0];
  const float* driving = (const float*)d_in[1];
  const float* W_in  = (const float*)d_in[2];
  const float* U_in  = (const float*)d_in[3];
  const float* v_in  = (const float*)d_in[4];
  const float* W_t   = (const float*)d_in[5];
  const float* U_t   = (const float*)d_in[6];
  const float* v_t   = (const float*)d_in[7];
  const float* Wih_e = (const float*)d_in[8];
  const float* Whh_e = (const float*)d_in[9];
  const float* bih_e = (const float*)d_in[10];
  const float* bhh_e = (const float*)d_in[11];
  const float* Wih_d = (const float*)d_in[12];
  const float* Whh_d = (const float*)d_in[13];
  const float* bih_d = (const float*)d_in[14];
  const float* bhh_d = (const float*)d_in[15];
  const float* Wy = (const float*)d_in[16];
  const float* by = (const float*)d_in[17];
  const float* Wo = (const float*)d_in[18];
  const float* bo = (const float*)d_in[19];
  float* out = (float*)d_out;

  // workspace layout
  const size_t PRE_IN_B = (size_t)B_ * N_ * T_ * 2;  // bf16, 134 MB (fits 192MB L2)
  const size_t DRVT_B   = (size_t)B_ * T_ * N_ * 2;  // bf16, 134 MB
  const size_t HOUT_B   = (size_t)B_ * T_ * H_ * 4;  // f32,   67 MB
  const size_t PREH_B   = (size_t)B_ * T_ * H_ * 4;  // f32,   67 MB
  if (ws_size < PRE_IN_B + DRVT_B + HOUT_B + PREH_B) return;

  char* ws = (char*)d_ws;
  __bf16* pre_in = (__bf16*)ws;
  __bf16* drvT   = (__bf16*)(ws + PRE_IN_B);
  float*  H_out  = (float*)(ws + PRE_IN_B + DRVT_B);
  float*  pre_h  = (float*)(ws + PRE_IN_B + DRVT_B + HOUT_B);

  // 1) driving -> [b][t][n] bf16
  k_transpose_drv<<<B_ * (N_ / 64), 256, 0, stream>>>(driving, drvT);
  // 2) pre_in = driving @ U_in  (bf16 out, L2-resident for the 64-step scan)
  k_gemm64<true><<<(B_ * N_) / 64, 128, 0, stream>>>(driving, U_in, pre_in);
  // 3) encoder scan -> H_out
  k_encoder<<<B_ / 16, 256, 0, stream>>>(pre_in, drvT, W_in, v_in, Wih_e, Whh_e,
                                         bih_e, bhh_e, H_out);
  // 4) pre_h = H_out @ U_t  (f32 out)
  k_gemm64<false><<<(B_ * T_) / 64, 128, 0, stream>>>(H_out, U_t, pre_h);
  // 5) decoder scan -> out
  k_decoder<<<B_ / 16, 256, 0, stream>>>(history, pre_h, H_out, W_t, v_t, Wih_d,
                                         Whh_d, bih_d, bhh_d, Wy, by, Wo, bo, out);
}
